// CrAKNAttention_80728205295731
// MI455X (gfx1250) — matrix operand order repaired and checked by reading, not compiled, not run on hardware
//
#include <hip/hip_runtime.h>
#include <hip/hip_bf16.h>
#include <math.h>

// Problem constants (from reference)
#define SEQ 768
#define DIN 256
#define NH  8
#define HDM 32
#define MDL 256   // NH*HDM

typedef __bf16 bf16x16 __attribute__((ext_vector_type(16)));
typedef float  f32x8   __attribute__((ext_vector_type(8)));
typedef float  f32x16  __attribute__((ext_vector_type(16)));

// ---------- helpers ----------
__device__ inline float mishf(float z) {
  // mish(z) = z * tanh(softplus(z)); tanh(log1p(e^z)) = ((1+e)^2-1)/((1+e)^2+1)
  float zz = fminf(z, 15.0f);
  float e  = __expf(zz);
  float n  = e * (e + 2.0f);          // (1+e)^2 - 1
  float t  = n / (n + 2.0f);
  return (z > 15.0f) ? z : z * t;
}

__device__ inline void put4(f32x16& v, int base, float4 a) {
  v[base + 0] = a.x; v[base + 1] = a.y; v[base + 2] = a.z; v[base + 3] = a.w;
}

// packed f32 -> bf16 conversion; lowers to v_cvt_pk_bf16_f32 pairs
__device__ inline bf16x16 cvt16(f32x16 f) {
  return __builtin_convertvector(f, bf16x16);
}

// ---------- generic GEMM: C[M,N] = act(A[M,K] @ W[N,K]^T + bias[N]) ----------
// one wave per 16x16 tile, K stepped by 32 via v_wmma_f32_16x16x32_bf16
template <int ACT>
__global__ __launch_bounds__(128)
void gemm_awt(const float* __restrict__ A, const float* __restrict__ W,
              const float* __restrict__ bias, float* __restrict__ C,
              int M, int N, int K) {
  const int lane = threadIdx.x;          // 0..31
  const int half = lane >> 4;            // 0/1
  const int r    = lane & 15;
  const int m0   = (blockIdx.y * blockDim.y + threadIdx.y) * 16;
  const int n0   = blockIdx.x * 16;

  const float* Arow = A + (size_t)(m0 + r) * K;   // A-frag row for this lane
  const float* Wrow = W + (size_t)(n0 + r) * K;   // B-frag column = W row n0+r

  f32x8 acc = {};
  for (int kb = 0; kb < K; kb += 32) {
    // A fragment: two contiguous 8-float chunks (half-dependent K offsets)
    const float4* ap0 = reinterpret_cast<const float4*>(Arow + kb + half * 16);
    const float4* ap1 = reinterpret_cast<const float4*>(Arow + kb + half * 16 + 8);
    f32x16 afv;
    put4(afv, 0, ap0[0]); put4(afv, 4,  ap0[1]);
    put4(afv, 8, ap1[0]); put4(afv, 12, ap1[1]);
    // B fragment: 16 contiguous floats of W row
    const float4* wp = reinterpret_cast<const float4*>(Wrow + kb + half * 16);
    f32x16 bfvf;
    put4(bfvf, 0, wp[0]); put4(bfvf, 4,  wp[1]);
    put4(bfvf, 8, wp[2]); put4(bfvf, 12, wp[3]);

    acc = __builtin_amdgcn_wmma_f32_16x16x32_bf16(
        false, cvt16(afv), false, cvt16(bfvf), (short)0, acc, false, false);
  }

  const float bv = bias ? bias[n0 + r] : 0.0f;
#pragma unroll
  for (int e = 0; e < 8; ++e) {
    float c = acc[e] + bv;
    if (ACT) c = mishf(c);
    C[(size_t)(m0 + half * 8 + e) * N + n0 + r] = c;
  }
}

// ---------- pair/scores kernel ----------
// scores[h, i, j] = q.k/sqrt(32) + sqrt(sum_e mish(G[j]-G[i]+b_de)^2) per head
__global__ __launch_bounds__(256)
void pair_scores(const float* __restrict__ G, const float* __restrict__ b_de,
                 const float* __restrict__ qkv, float* __restrict__ scores) {
  const int lane  = threadIdx.x & 31;
  const int wave  = threadIdx.x >> 5;           // 0..7
  const int i     = blockIdx.x;
  const int jbase = blockIdx.y * 64 + wave * 8;
  const int c0    = lane * 8;                   // model-channel base (8 per lane)
  const int h     = lane >> 2;                  // head served by this lane group
  const int e0    = (lane & 3) * 8;             // within-head element base

  // per-block constants for row i
  float gi[8], bde[8], qv[8];
  {
    const float4* g4 = reinterpret_cast<const float4*>(G + (size_t)i * MDL + c0);
    float4 ga = g4[0], gb = g4[1];
    gi[0]=ga.x; gi[1]=ga.y; gi[2]=ga.z; gi[3]=ga.w;
    gi[4]=gb.x; gi[5]=gb.y; gi[6]=gb.z; gi[7]=gb.w;
    const float4* b4 = reinterpret_cast<const float4*>(b_de + c0);
    float4 ba = b4[0], bb = b4[1];
    bde[0]=ba.x; bde[1]=ba.y; bde[2]=ba.z; bde[3]=ba.w;
    bde[4]=bb.x; bde[5]=bb.y; bde[6]=bb.z; bde[7]=bb.w;
    const float4* q4 = reinterpret_cast<const float4*>(qkv + (size_t)i * (3*MDL) + h * 96 + e0);
    float4 qa = q4[0], qb = q4[1];
    qv[0]=qa.x; qv[1]=qa.y; qv[2]=qa.z; qv[3]=qa.w;
    qv[4]=qb.x; qv[5]=qb.y; qv[6]=qb.z; qv[7]=qb.w;
  }

  const float rs32 = 0.17677669529663687f;  // 1/sqrt(32)

#pragma unroll 1
  for (int jj = 0; jj < 8; ++jj) {
    const int j = jbase + jj;
    const float4* g4 = reinterpret_cast<const float4*>(G + (size_t)j * MDL + c0);
    float4 ga = g4[0], gb = g4[1];
    const float4* k4 = reinterpret_cast<const float4*>(qkv + (size_t)j * (3*MDL) + h * 96 + 32 + e0);
    float4 ka = k4[0], kb4 = k4[1];

    float gj[8] = { ga.x, ga.y, ga.z, ga.w, gb.x, gb.y, gb.z, gb.w };
    float kv[8] = { ka.x, ka.y, ka.z, ka.w, kb4.x, kb4.y, kb4.z, kb4.w };

    float accd = 0.0f, accq = 0.0f;
#pragma unroll
    for (int t = 0; t < 8; ++t) {
      float y = mishf(gj[t] - gi[t] + bde[t]);
      accd = fmaf(y, y, accd);
      accq = fmaf(qv[t], kv[t], accq);
    }
    // reduce across the 4 lanes of this head group
    accd += __shfl_xor(accd, 1, 32);  accd += __shfl_xor(accd, 2, 32);
    accq += __shfl_xor(accq, 1, 32);  accq += __shfl_xor(accq, 2, 32);

    if ((lane & 3) == 0) {
      scores[((size_t)h * SEQ + i) * SEQ + j] = accq * rs32 + sqrtf(accd);
    }
  }
}

// ---------- row softmax over j (in place) ----------
__global__ __launch_bounds__(256)
void softmax_rows(float* __restrict__ scores) {
  const int tid = threadIdx.x;
  float* row = scores + (size_t)blockIdx.x * SEQ;
  __shared__ float red[256];

  float v0 = row[tid], v1 = row[tid + 256], v2 = row[tid + 512];
  float m = fmaxf(v0, fmaxf(v1, v2));
  red[tid] = m; __syncthreads();
  for (int s = 128; s > 0; s >>= 1) {
    if (tid < s) red[tid] = fmaxf(red[tid], red[tid + s]);
    __syncthreads();
  }
  const float rowmax = red[0]; __syncthreads();

  float e0 = __expf(v0 - rowmax), e1 = __expf(v1 - rowmax), e2 = __expf(v2 - rowmax);
  red[tid] = e0 + e1 + e2; __syncthreads();
  for (int s = 128; s > 0; s >>= 1) {
    if (tid < s) red[tid] += red[tid + s];
    __syncthreads();
  }
  const float inv = 1.0f / red[0];
  row[tid] = e0 * inv; row[tid + 256] = e1 * inv; row[tid + 512] = e2 * inv;
}

// ---------- attn @ v (B not transposed, column-strided) ----------
__global__ __launch_bounds__(128)
void av_gemm(const float* __restrict__ attn, const float* __restrict__ qkv,
             float* __restrict__ vals) {
  const int lane = threadIdx.x;
  const int half = lane >> 4;
  const int r    = lane & 15;
  const int head = blockIdx.z;
  const int m0   = (blockIdx.y * blockDim.y + threadIdx.y) * 16;
  const int n0   = blockIdx.x * 16;

  const float* Arow = attn + ((size_t)head * SEQ + m0 + r) * SEQ;
  const float* Bcol = qkv + head * 96 + 64 + n0 + r;    // v[h, k, n]: + k*(3M)

  f32x8 acc = {};
  for (int kb = 0; kb < SEQ; kb += 32) {
    const float4* ap0 = reinterpret_cast<const float4*>(Arow + kb + half * 16);
    const float4* ap1 = reinterpret_cast<const float4*>(Arow + kb + half * 16 + 8);
    f32x16 afv;
    put4(afv, 0, ap0[0]); put4(afv, 4,  ap0[1]);
    put4(afv, 8, ap1[0]); put4(afv, 12, ap1[1]);

    f32x16 bfvf;
#pragma unroll
    for (int e = 0; e < 16; ++e)
      bfvf[e] = Bcol[(size_t)(kb + half * 16 + e) * (3*MDL)];

    acc = __builtin_amdgcn_wmma_f32_16x16x32_bf16(
        false, cvt16(afv), false, cvt16(bfvf), (short)0, acc, false, false);
  }
#pragma unroll
  for (int e = 0; e < 8; ++e)
    vals[(size_t)(m0 + half * 8 + e) * MDL + head * HDM + n0 + r] = acc[e];
}

// ---------- host launcher ----------
extern "C" void kernel_launch(void* const* d_in, const int* in_sizes, int n_in,
                              void* d_out, int out_size, void* d_ws, size_t ws_size,
                              hipStream_t stream) {
  const float* x     = (const float*)d_in[0];
  const float* bias  = (const float*)d_in[1];
  const float* W_qkv = (const float*)d_in[2];
  const float* b_qkv = (const float*)d_in[3];
  const float* W_be  = (const float*)d_in[4];
  const float* b_be  = (const float*)d_in[5];
  const float* W_de  = (const float*)d_in[6];
  const float* b_de  = (const float*)d_in[7];
  const float* W_o   = (const float*)d_in[8];
  const float* b_o   = (const float*)d_in[9];
  const float* W_bo  = (const float*)d_in[10];
  const float* b_bo  = (const float*)d_in[11];

  float* out      = (float*)d_out;                 // [768,256]
  float* bias_out = out + (size_t)SEQ * DIN;       // [768,256]

  float* ws     = (float*)d_ws;
  float* qkv    = ws;                              // 768*768
  float* bias_e = qkv    + (size_t)SEQ * 3 * MDL;  // 768*256
  float* G      = bias_e + (size_t)SEQ * MDL;      // 768*256
  float* vals   = G      + (size_t)SEQ * MDL;      // 768*256
  float* scores = vals   + (size_t)SEQ * MDL;      // 8*768*768

  dim3 blk(32, 4);

  // qkv = x @ W_qkv^T + b_qkv            [768 x 768], K=256
  gemm_awt<0><<<dim3(48, 12), blk, 0, stream>>>(x, W_qkv, b_qkv, qkv, SEQ, 3*MDL, DIN);
  // bias_e = mish(bias @ W_be^T + b_be)  [768 x 256], K=256
  gemm_awt<1><<<dim3(16, 12), blk, 0, stream>>>(bias, W_be, b_be, bias_e, SEQ, MDL, DIN);
  // G = bias_e @ W_de^T (b_de applied per pair)
  gemm_awt<0><<<dim3(16, 12), blk, 0, stream>>>(bias_e, W_de, nullptr, G, SEQ, MDL, MDL);
  // bias_out = mish(bias_e @ W_bo^T + b_bo)
  gemm_awt<1><<<dim3(16, 12), blk, 0, stream>>>(bias_e, W_bo, b_bo, bias_out, SEQ, DIN, MDL);

  // scores[h,i,j] = q.k/sqrt(32) + ||mish(G[j]-G[i]+b_de)||_head
  pair_scores<<<dim3(SEQ, 12), 256, 0, stream>>>(G, b_de, qkv, scores);

  // softmax over j, in place
  softmax_rows<<<dim3(NH * SEQ), 256, 0, stream>>>(scores);

  // vals = attn @ v  (per head), laid out [S, M]
  av_gemm<<<dim3(2, 12, NH), blk, 0, stream>>>(scores, qkv, vals);

  // out = vals @ W_o^T + b_o
  gemm_awt<0><<<dim3(16, 12), blk, 0, stream>>>(vals, W_o, b_o, out, SEQ, DIN, MDL);
}
